// xLSTMMoELayer_67207648248430
// MI455X (gfx1250) — matrix operand order repaired and checked by reading, not compiled, not run on hardware
//
#include <hip/hip_runtime.h>
#include <cstdint>
#include <cstddef>

// Problem constants (from reference)
constexpr int NB   = 4;
constexpr int NS   = 2048;
constexpr int NTOK = NB * NS;   // 8192 tokens
constexpr int DD   = 1024;      // model dim
constexpr int FF   = 4096;      // expert hidden dim
constexpr int NE   = 8;         // experts
constexpr int TK   = 2;         // top-k

typedef __attribute__((ext_vector_type(16))) __bf16 v16bf;
typedef __attribute__((ext_vector_type(8)))  float  v8f;
typedef unsigned int u32x4 __attribute__((ext_vector_type(4)));
typedef int          i32x4 __attribute__((ext_vector_type(4)));
typedef int          i32x8 __attribute__((ext_vector_type(8)));

union FragAB { v16bf v; uint4 q[2]; };

#if defined(__has_builtin)
#  if __has_builtin(__builtin_amdgcn_tensor_load_to_lds)
#    define HAVE_TDM 1
#  endif
#endif
#ifndef HAVE_TDM
#  define HAVE_TDM 0
#endif

__device__ __forceinline__ unsigned short f32_to_bf16(float f) {
  unsigned u = __float_as_uint(f);
  u += 0x7FFFu + ((u >> 16) & 1u);            // round-to-nearest-even
  return (unsigned short)(u >> 16);
}

__device__ __forceinline__ float gelu_tanh(float x) {
  const float c = 0.7978845608028654f;        // sqrt(2/pi)
  float t = tanhf(c * (x + 0.044715f * x * x * x));
  return 0.5f * x * (1.0f + t);
}

// ---- CDNA5 async global->LDS (per-lane addresses), ISA §15.18 opcode 98
__device__ __forceinline__ void async_ld_b128(unsigned int lds_off, const void* gsrc) {
  unsigned long long g = (unsigned long long)(uintptr_t)gsrc;
  asm volatile("global_load_async_to_lds_b128 %0, %1, off" :: "v"(lds_off), "v"(g) : "memory");
}
__device__ __forceinline__ void wait_async0() {
#if defined(__has_builtin) && __has_builtin(__builtin_amdgcn_s_wait_asynccnt)
  __builtin_amdgcn_s_wait_asynccnt(0);
#else
  asm volatile("s_wait_asynccnt 0" ::: "memory");
#endif
}
__device__ __forceinline__ void wait_tensor0() {
#if defined(__has_builtin) && __has_builtin(__builtin_amdgcn_s_wait_tensorcnt)
  __builtin_amdgcn_s_wait_tensorcnt(0);
#else
  asm volatile("s_wait_tensorcnt 0" ::: "memory");
#endif
}

#if HAVE_TDM
// ---- TDM 2D tile load with LDS padding: 128B rows -> +16B pad (pitch 144B)
// D# per CDNA5 ISA. data_size=1 (2 bytes/elem).
__device__ __forceinline__ void tdm_load_tile_2d(unsigned int lds_off, const void* gaddr,
                                                 unsigned int tile_w, unsigned int tile_h,
                                                 unsigned int stride_elems,
                                                 unsigned int tens_w, unsigned int tens_h) {
  unsigned long long ga = (unsigned long long)(uintptr_t)gaddr;
  u32x4 g0;
  g0[0] = 1u;                                                    // count=1, user mode
  g0[1] = lds_off;                                               // lds_addr [63:32]
  g0[2] = (unsigned int)ga;                                      // global_addr [95:64]
  g0[3] = (unsigned int)((ga >> 32) & 0x01FFFFFFu) | (2u << 30); // addr[56:32] | type=2
  i32x8 g1;
  // data_size=1(2B) | pad_enable | pad_interval=4 (32 DW=128B) | pad_amount=3 (4 DW=16B)
  g1[0] = (int)((1u << 16) | (1u << 20) | (4u << 22) | (3u << 25));
  g1[1] = (int)((tens_w & 0xFFFFu) << 16);                       // tensor_dim0 [79:48]
  g1[2] = (int)(((tens_w >> 16) & 0xFFFFu) | ((tens_h & 0xFFFFu) << 16)); // dim0 hi | dim1 lo
  g1[3] = (int)(((tens_h >> 16) & 0xFFFFu) | ((tile_w & 0xFFFFu) << 16)); // dim1 hi | tile_dim0
  g1[4] = (int)(tile_h & 0xFFFFu);                               // tile_dim1 (tile_dim2=0)
  g1[5] = (int)stride_elems;                                     // tensor_dim0_stride lo32
  g1[6] = 0;
  g1[7] = 0;
  i32x4 z4 = {0, 0, 0, 0};
#if defined(__has_include)
#  if __has_include(<hip/amd_detail/amd_gfx1250_TDM.h>)
  i32x8 z8 = {0, 0, 0, 0, 0, 0, 0, 0};
  __builtin_amdgcn_tensor_load_to_lds(g0, g1, z4, z4, z8, 0);    // clang-23 6-arg form
#  else
  __builtin_amdgcn_tensor_load_to_lds(g0, g1, z4, z4, 0);        // ROCm 7.2 5-arg form
#  endif
#else
  __builtin_amdgcn_tensor_load_to_lds(g0, g1, z4, z4, 0);
#endif
}
#endif

// ---------------------------------------------------------------- utilities
__global__ void init_kernel(int* counts, unsigned short* zrow) {
  int t = threadIdx.x;
  if (t < NE) counts[t] = 0;
  if (t < 64) zrow[t] = 0;     // zero bf16 row for invalid gather slots
}

__global__ void cvt_f32_bf16_kernel(const float* __restrict__ in,
                                    unsigned short* __restrict__ out, size_t n) {
  size_t i = (size_t)blockIdx.x * blockDim.x + threadIdx.x;
  size_t stride = (size_t)gridDim.x * blockDim.x;
  for (; i < n; i += stride) out[i] = f32_to_bf16(in[i]);
}

// src [E][R][C] f32 -> dst [E][C][R] bf16, 32x32 LDS tiles
__global__ __launch_bounds__(256)
void transpose_cvt_kernel(const float* __restrict__ src, unsigned short* __restrict__ dst,
                          int R, int C) {
  __shared__ float tile[32][33];
  int tilesR = R >> 5, tilesC = C >> 5;
  int bid = blockIdx.x;
  int e   = bid / (tilesR * tilesC);
  int rem = bid % (tilesR * tilesC);
  int tr  = rem / tilesC, tc = rem % tilesC;
  const float* s = src + ((size_t)e * R + tr * 32) * C + tc * 32;
  unsigned short* d = dst + ((size_t)e * C + tc * 32) * R + tr * 32;
#pragma unroll
  for (int it = 0; it < 4; ++it) {
    int idx = threadIdx.x + it * 256;
    int r = idx >> 5, c = idx & 31;
    tile[r][c] = s[(size_t)r * C + c];
  }
  __syncthreads();
#pragma unroll
  for (int it = 0; it < 4; ++it) {
    int idx = threadIdx.x + it * 256;
    int r = idx >> 5, c = idx & 31;
    d[(size_t)r * R + c] = f32_to_bf16(tile[c][r]);
  }
}

// ---------------------------------------------------------------- router
__global__ __launch_bounds__(256)
void router_kernel(const float* __restrict__ x, const float* __restrict__ gw,
                   const float* __restrict__ gb, float* __restrict__ topw,
                   int* __restrict__ counts, int* __restrict__ tok_list,
                   int* __restrict__ slot_list) {
  int t = blockIdx.x * blockDim.x + threadIdx.x;
  if (t >= NTOK) return;
  float acc[NE];
#pragma unroll
  for (int e = 0; e < NE; ++e) acc[e] = gb[e];
  const float* xr = x + (size_t)t * DD;
  for (int d = 0; d < DD; ++d) {
    float xv = xr[d];
    const float* g = gw + (size_t)d * NE;
#pragma unroll
    for (int e = 0; e < NE; ++e) acc[e] += xv * g[e];
  }
  float m = acc[0];
#pragma unroll
  for (int e = 1; e < NE; ++e) m = fmaxf(m, acc[e]);
  float p[NE];
#pragma unroll
  for (int e = 0; e < NE; ++e) p[e] = expf(acc[e] - m);
  int i0 = 0;
#pragma unroll
  for (int e = 1; e < NE; ++e) if (p[e] > p[i0]) i0 = e;   // first index wins ties
  int i1 = (i0 == 0) ? 1 : 0;
#pragma unroll
  for (int e = 0; e < NE; ++e) if (e != i0 && p[e] > p[i1]) i1 = e;
  float w0 = p[i0], w1 = p[i1], ws = w0 + w1;              // softmax denom cancels
  topw[t * TK + 0] = w0 / ws;
  topw[t * TK + 1] = w1 / ws;
  int p0 = atomicAdd(&counts[i0], 1);
  tok_list[i0 * NTOK + p0]  = t;
  slot_list[i0 * NTOK + p0] = t * TK + 0;
  int p1 = atomicAdd(&counts[i1], 1);
  tok_list[i1 * NTOK + p1]  = t;
  slot_list[i1 * NTOK + p1] = t * TK + 1;
}

// ---------------------------------------------------------------- gathered GEMM
// Transposed-output formulation with DOUBLE-BUFFERED async/TDM staging:
//   stage s+1 TDM + async loads fly while WMMAs consume stage s.
// Fragment layouts (b128 LDS reads, 144B row pitch):
//   A 16x32 bf16: lanes 0-15: M=lane, K{0..7,16..23}; lanes 16-31: K{8..15,24..31}
//   B 32x16 bf16: lanes 0-15: N=lane, K{0..7,16..23}; lanes 16-31: K{8..15,24..31}
//   C 16x16 f32 : VGPR i -> M = i + 8*(lane>=16), N = lane&15
template<int KD, int ND, bool DO_GELU, bool OUT_BF16>
__global__ __launch_bounds__(256)
void moe_gemm_kernel(const unsigned short* __restrict__ Xbase, // token rows [*, KD] bf16
                     const unsigned short* __restrict__ Wt,    // [NE][ND][KD] bf16 (pre-transposed)
                     const float* __restrict__ bias,           // [NE][ND]
                     const int* __restrict__ counts,
                     const int* __restrict__ rowmap,           // [NE][NTOK] X-row index
                     const int* __restrict__ slotmap,          // [NE][NTOK] out-row index
                     const unsigned short* __restrict__ zrow,  // 64 zero bf16
                     void* __restrict__ outv) {                // [NTOK*TK][ND]
  constexpr int BFD = 128, BT = 128, BK = 64, LDK = BK + 8;    // 144B LDS row pitch
  constexpr int NT = ND / BFD;
  constexpr int MT = NTOK / BT;
  constexpr unsigned int BUFB = (unsigned int)(BFD * LDK * 2); // bytes per LDS buffer

  const int bid = blockIdx.x;
  const int e   = bid / (MT * NT);
  const int rem = bid % (MT * NT);
  const int tt  = rem / NT;                  // token tile
  const int tf  = rem % NT;                  // out-col tile
  const int cnt = counts[e];
  const int row0 = tt * BT;
  if (row0 >= cnt) return;                   // uniform: whole block exits
  const int fB0 = tf * BFD;

  __shared__ __align__(16) unsigned short Ws[2][BFD][LDK];
  __shared__ __align__(16) unsigned short Xs[2][BT][LDK];

  const int tid  = threadIdx.x;
  const int lane = tid & 31, wv = tid >> 5;
  const int wm = wv >> 2, wn = wv & 3;       // 2x4 wave grid -> 64(out) x 32(tok) per wave
  const int half = lane >> 4, l15 = lane & 15;

  const unsigned short* We = Wt + ((size_t)e * ND + fB0) * KD;
  const int* rmap = rowmap  + (size_t)e * NTOK;
  const int* smap = slotmap + (size_t)e * NTOK;

  // Per-thread X gather plan: 4 x 16B chunks per k-stage; invalid rows read zrow.
  const unsigned short* xsrc[4];
  unsigned int xstep[4];
  unsigned int xdst[4];
#pragma unroll
  for (int it = 0; it < 4; ++it) {
    int c  = tid + it * 256;                 // chunk id 0..1023
    int r  = c >> 3;                         // row 0..127
    int kc = (c & 7) * 8;                    // k-chunk (8 bf16 = 16B)
    int pos = row0 + r;
    if (pos < cnt) { xsrc[it] = Xbase + (size_t)rmap[pos] * KD + kc; xstep[it] = BK; }
    else           { xsrc[it] = zrow + kc;                           xstep[it] = 0;  }
    xdst[it] = (unsigned int)(uintptr_t)&Xs[0][r][kc];
  }
  const unsigned int ws_lds0 = (unsigned int)(uintptr_t)&Ws[0][0][0];

  // Issue all loads for one k-stage into LDS buffer `buf` (consumes xsrc cursor).
  auto issue_stage = [&](int k0, int buf) {
#if HAVE_TDM
    if (wv == 0) {                           // one TDM op stages the whole W tile
      tdm_load_tile_2d(ws_lds0 + (unsigned int)buf * BUFB, We + k0, BK, BFD, KD, KD, ND);
    }
#else
#pragma unroll
    for (int it = 0; it < 4; ++it) {
      int c  = tid + it * 256;
      int fr = c >> 3;
      int kc = (c & 7) * 8;
      async_ld_b128((unsigned int)(uintptr_t)&Ws[buf][fr][kc],
                    We + (size_t)fr * KD + k0 + kc);
    }
#endif
#pragma unroll
    for (int it = 0; it < 4; ++it) {
      async_ld_b128(xdst[it] + (unsigned int)buf * BUFB, xsrc[it]);
      xsrc[it] += xstep[it];
    }
  };

  v8f acc[4][2] = {};
  constexpr int NST = KD / BK;

  issue_stage(0, 0);                         // prologue: stage 0 in flight

  for (int s = 0; s < NST; ++s) {
    const int buf = s & 1;
    wait_async0();                           // stage s loads (issued a full stage ago)
    wait_tensor0();
    __syncthreads();                         // stage s visible; buf^1 free to overwrite
    if (s + 1 < NST) issue_stage((s + 1) * BK, buf ^ 1);   // overlap with compute below
    if (s + 2 < NST) {                       // L2 prefetch one stage beyond in-flight TDM
      __builtin_prefetch(We + (size_t)(tid >> 1) * KD + (s + 2) * BK + (tid & 1) * 32, 0, 1);
    }

#pragma unroll
    for (int kk = 0; kk < BK; kk += 32) {
      FragAB am[4], bn[2];
#pragma unroll
      for (int mt = 0; mt < 4; ++mt) {
        int r = wm * 64 + mt * 16 + l15;
        am[mt].q[0] = *(const uint4*)&Ws[buf][r][kk + half * 8];
        am[mt].q[1] = *(const uint4*)&Ws[buf][r][kk + 16 + half * 8];
      }
#pragma unroll
      for (int nt = 0; nt < 2; ++nt) {
        int t = wn * 32 + nt * 16 + l15;
        bn[nt].q[0] = *(const uint4*)&Xs[buf][t][kk + half * 8];
        bn[nt].q[1] = *(const uint4*)&Xs[buf][t][kk + 16 + half * 8];
      }
#pragma unroll
      for (int mt = 0; mt < 4; ++mt)
#pragma unroll
        for (int nt = 0; nt < 2; ++nt)
          acc[mt][nt] = __builtin_amdgcn_wmma_f32_16x16x32_bf16(
              false, am[mt].v, false, bn[nt].v, (short)0, acc[mt][nt], false, false);
    }
  }

  // ---- epilogue: per (mt,nt) each thread owns 8 CONSECUTIVE out-cols of one out-row
#pragma unroll
  for (int nt = 0; nt < 2; ++nt) {
    int pos = row0 + wn * 32 + nt * 16 + l15;
    bool valid = pos < cnt;
    int slot = valid ? smap[pos] : 0;
#pragma unroll
    for (int mt = 0; mt < 4; ++mt) {
      int f0 = fB0 + wm * 64 + mt * 16 + half * 8;   // 8 consecutive out-cols
      float4 blo = *(const float4*)&bias[(size_t)e * ND + f0];
      float4 bhi = *(const float4*)&bias[(size_t)e * ND + f0 + 4];
      float v[8];
#pragma unroll
      for (int i = 0; i < 8; ++i) {
        float bv = (i < 4) ? (&blo.x)[i] : (&bhi.x)[i - 4];
        float x = acc[mt][nt][i] + bv;
        v[i] = DO_GELU ? gelu_tanh(x) : x;
      }
      if (valid) {
        if (OUT_BF16) {
          uint4 pk;
          pk.x = (unsigned)f32_to_bf16(v[0]) | ((unsigned)f32_to_bf16(v[1]) << 16);
          pk.y = (unsigned)f32_to_bf16(v[2]) | ((unsigned)f32_to_bf16(v[3]) << 16);
          pk.z = (unsigned)f32_to_bf16(v[4]) | ((unsigned)f32_to_bf16(v[5]) << 16);
          pk.w = (unsigned)f32_to_bf16(v[6]) | ((unsigned)f32_to_bf16(v[7]) << 16);
          *(uint4*)((unsigned short*)outv + (size_t)slot * ND + f0) = pk;
        } else {
          float* op = (float*)outv + (size_t)slot * ND + f0;
          *(float4*)op       = make_float4(v[0], v[1], v[2], v[3]);
          *(float4*)(op + 4) = make_float4(v[4], v[5], v[6], v[7]);
        }
      }
    }
  }
}

// ---------------------------------------------------------------- combine
__global__ void combine_kernel(const float* __restrict__ y, const float* __restrict__ topw,
                               float* __restrict__ out) {
  size_t i = (size_t)blockIdx.x * blockDim.x + threadIdx.x;
  size_t n = (size_t)NTOK * DD;
  size_t stride = (size_t)gridDim.x * blockDim.x;
  for (; i < n; i += stride) {
    size_t t = i / DD, d = i % DD;
    out[i] = topw[t * TK + 0] * y[(t * TK + 0) * (size_t)DD + d]
           + topw[t * TK + 1] * y[(t * TK + 1) * (size_t)DD + d];
  }
}

// ---------------------------------------------------------------- launch
extern "C" void kernel_launch(void* const* d_in, const int* in_sizes, int n_in,
                              void* d_out, int out_size, void* d_ws, size_t ws_size,
                              hipStream_t stream) {
  const float* h_t    = (const float*)d_in[0];   // [NTOK, DD]
  const float* gate_w = (const float*)d_in[1];   // [DD, NE]
  const float* gate_b = (const float*)d_in[2];   // [NE]
  const float* w1     = (const float*)d_in[3];   // [NE, DD, FF]
  const float* b1     = (const float*)d_in[4];   // [NE, FF]
  const float* w2     = (const float*)d_in[5];   // [NE, FF, DD]
  const float* b2     = (const float*)d_in[6];   // [NE, DD]
  float* out = (float*)d_out;

  char* ws = (char*)d_ws;
  size_t off = 0;
  auto carve = [&](size_t bytes) { size_t o = off; off += (bytes + 255) & ~(size_t)255; return o; };
  size_t xb_o   = carve((size_t)NTOK * DD * 2);          // bf16 tokens
  size_t w1t_o  = carve((size_t)NE * FF * DD * 2);       // bf16 W1^T  [E][FF][DD]
  size_t w2t_o  = carve((size_t)NE * DD * FF * 2);       // bf16 W2^T  [E][DD][FF]
  size_t hid_o  = carve((size_t)NTOK * TK * FF * 2);     // bf16 hidden per (token,k)
  size_t y_o    = carve((size_t)NTOK * TK * DD * 4);     // f32 expert outputs
  size_t topw_o = carve((size_t)NTOK * TK * 4);
  size_t cnt_o  = carve((size_t)NE * 4);
  size_t tok_o  = carve((size_t)NE * NTOK * 4);
  size_t slot_o = carve((size_t)NE * NTOK * 4);
  size_t zrow_o = carve((size_t)64 * 2);
  (void)ws_size; (void)in_sizes; (void)n_in; (void)out_size;

  unsigned short* xb   = (unsigned short*)(ws + xb_o);
  unsigned short* w1t  = (unsigned short*)(ws + w1t_o);
  unsigned short* w2t  = (unsigned short*)(ws + w2t_o);
  unsigned short* hid  = (unsigned short*)(ws + hid_o);
  float* yb   = (float*)(ws + y_o);
  float* topw = (float*)(ws + topw_o);
  int* counts = (int*)(ws + cnt_o);
  int* tokl   = (int*)(ws + tok_o);
  int* slotl  = (int*)(ws + slot_o);
  unsigned short* zrow = (unsigned short*)(ws + zrow_o);

  init_kernel<<<1, 64, 0, stream>>>(counts, zrow);

  cvt_f32_bf16_kernel<<<2048, 256, 0, stream>>>(h_t, xb, (size_t)NTOK * DD);
  transpose_cvt_kernel<<<NE * (DD / 32) * (FF / 32), 256, 0, stream>>>(w1, w1t, DD, FF);
  transpose_cvt_kernel<<<NE * (FF / 32) * (DD / 32), 256, 0, stream>>>(w2, w2t, FF, DD);

  router_kernel<<<NTOK / 256, 256, 0, stream>>>(h_t, gate_w, gate_b,
                                                topw, counts, tokl, slotl);

  // GEMM1: hid[slot][f] = gelu(x[tok] . w1t[e][f] + b1[e][f])
  moe_gemm_kernel<DD, FF, true, true>
      <<<NE * (NTOK / 128) * (FF / 128), 256, 0, stream>>>(
          xb, w1t, b1, counts, tokl, slotl, zrow, (void*)hid);

  // GEMM2: y[slot][d] = hid[slot] . w2t[e][d] + b2[e][d]
  moe_gemm_kernel<FF, DD, false, false>
      <<<NE * (NTOK / 128) * (DD / 128), 256, 0, stream>>>(
          hid, w2t, b2, counts, slotl, slotl, zrow, (void*)yb);

  combine_kernel<<<4096, 256, 0, stream>>>(yb, topw, out);
}
